// MegablockMoE_15925738733962
// MI455X (gfx1250) — compile-verified, matrix operand order checked by reading, not run on hardware
//
#include <hip/hip_runtime.h>
#include <hip/hip_bf16.h>
#include <math.h>

// ---- problem constants (from reference) ----
#define BB 4
#define SS 2048
#define DD 1024
#define EE 8
#define KK 2
#define DFF 4096
#define TT (BB * SS)          // 8192 tokens
#define CAP 2048              // expert capacity C = K*T/E
#define NSLOT (EE * CAP)      // 16384

typedef __bf16 bf16;
typedef __attribute__((ext_vector_type(16))) __bf16 v16bf;
typedef __attribute__((ext_vector_type(8)))  __bf16 v8bf;
typedef __attribute__((ext_vector_type(8)))  float  v8f;

// LDS tile geometry: K-tile 64 elements, padded row stride 72 (144B -> 16
// distinct banks across the 16 fragment-reading lanes).
#define KTILE 64
#define LDSPITCH 72
#define TILE_ELEMS (128 * LDSPITCH)          // one 128-row tile (A or B)
#define STAGE_ELEMS (2 * TILE_ELEMS)         // A tile + B tile

// ---------------------------------------------------------------------
// async global -> LDS copy (CDNA5 GLOBAL_LOAD_ASYNC_TO_LDS_B128, ASYNCcnt)
// ---------------------------------------------------------------------
__device__ __forceinline__ void async_copy_b128(const void* gptr, void* lptr) {
  const unsigned ldsoff = (unsigned)(uintptr_t)lptr;  // low 32 bits = LDS byte offset
  asm volatile("global_load_async_to_lds_b128 %0, %1, off"
               :: "v"(ldsoff), "v"(gptr) : "memory");
}

__device__ __forceinline__ void wait_async_zero() {
#if defined(__has_builtin)
#if __has_builtin(__builtin_amdgcn_s_wait_asynccnt)
  __builtin_amdgcn_s_wait_asynccnt(0);
#else
  asm volatile("s_wait_asynccnt 0" ::: "memory");
#endif
#else
  asm volatile("s_wait_asynccnt 0" ::: "memory");
#endif
}

// stage a 128 x 64 bf16 tile (row stride ld elements) into padded LDS.
// Uniform, fully unrolled trip count -> no EXEC manipulation around the
// async issues (4 chunks per thread, 256 threads, 1024 x 16B chunks).
__device__ __forceinline__ void stage_tile128(const bf16* __restrict__ g, size_t ld,
                                              bf16* lds, int tid) {
#pragma unroll
  for (int i = 0; i < 4; ++i) {
    const int c = tid + i * 256;
    const int row = c >> 3, col = c & 7;
    async_copy_b128(g + (size_t)row * ld + col * 8, lds + row * LDSPITCH + col * 8);
  }
}

// =====================================================================
// K0: tiled transpose + fp32->bf16 cast.  in: [E][R][Cc] f32, out: [E][Cc][R] bf16
// =====================================================================
__global__ void moe_transpose_cast(const float* __restrict__ in,
                                   bf16* __restrict__ out, int R, int Cc) {
  __shared__ float tile[32][33];
  const int e = blockIdx.z;
  const int c0 = blockIdx.x * 32;
  const int r0 = blockIdx.y * 32;
  in  += (size_t)e * R * Cc;
  out += (size_t)e * Cc * R;
  const int x = threadIdx.x, y = threadIdx.y;
#pragma unroll
  for (int k = 0; k < 4; ++k)
    tile[y + 8 * k][x] = in[(size_t)(r0 + y + 8 * k) * Cc + (c0 + x)];
  __syncthreads();
#pragma unroll
  for (int k = 0; k < 4; ++k)
    out[(size_t)(c0 + y + 8 * k) * R + (r0 + x)] = (bf16)tile[x][y + 8 * k];
}

// =====================================================================
// K1: router.  one wave per token; softmax over E=8; top-2 (first-index ties).
// token t -> x[b][s], b = t % B, s = t / B (megablocks layout).
// =====================================================================
__global__ void moe_router(const float* __restrict__ x,
                           const float* __restrict__ wr,
                           float* __restrict__ ew, int* __restrict__ eidx) {
  const int wave = threadIdx.x >> 5;
  const int lane = threadIdx.x & 31;
  const int t = blockIdx.x * 8 + wave;
  const int b = t & (BB - 1);
  const int s = t >> 2;
  const float* xr = x + ((size_t)b * SS + s) * DD;

  float acc[EE];
#pragma unroll
  for (int e = 0; e < EE; ++e) acc[e] = 0.f;
  for (int d = lane; d < DD; d += 32) {
    const float xv = xr[d];
    const float* w = wr + (size_t)d * EE;
#pragma unroll
    for (int e = 0; e < EE; ++e) acc[e] += xv * w[e];
  }
#pragma unroll
  for (int off = 16; off > 0; off >>= 1)
#pragma unroll
    for (int e = 0; e < EE; ++e) acc[e] += __shfl_down(acc[e], off, 32);

  if (lane == 0) {
    float m = acc[0];
#pragma unroll
    for (int e = 1; e < EE; ++e) m = fmaxf(m, acc[e]);
    float sum = 0.f, p[EE];
#pragma unroll
    for (int e = 0; e < EE; ++e) { p[e] = __expf(acc[e] - m); sum += p[e]; }
    const float inv = 1.f / sum;
#pragma unroll
    for (int e = 0; e < EE; ++e) p[e] *= inv;
    int i1 = 0; float v1 = p[0];
#pragma unroll
    for (int e = 1; e < EE; ++e) if (p[e] > v1) { v1 = p[e]; i1 = e; }
    int i2 = (i1 == 0) ? 1 : 0; float v2 = p[i2];
#pragma unroll
    for (int e = 0; e < EE; ++e)
      if (e != i1 && p[e] > v2) { v2 = p[e]; i2 = e; }
    ew[t * 2 + 0] = v1; eidx[t * 2 + 0] = i1;
    ew[t * 2 + 1] = v2; eidx[t * 2 + 1] = i2;
  }
}

// =====================================================================
// K2: stable sort-by-expert ranking + capacity drop (single workgroup).
// pos[f] = #{g < f : e_g == e_f}; keep = pos < C.
// =====================================================================
__global__ __launch_bounds__(1024)
void moe_assign(const int* __restrict__ eidx, const float* __restrict__ ew,
                int* __restrict__ slot, int* __restrict__ slot2token,
                float* __restrict__ slot2w) {
  __shared__ int cnt[1024 * EE];
  const int tid = threadIdx.x;

#pragma unroll
  for (int i = 0; i < 16; ++i) slot2token[tid * 16 + i] = -1;

  int lc[EE];
#pragma unroll
  for (int e = 0; e < EE; ++e) lc[e] = 0;
  int le[16];
#pragma unroll
  for (int i = 0; i < 16; ++i) {
    le[i] = eidx[tid * 16 + i];
    lc[le[i]]++;
  }
#pragma unroll
  for (int e = 0; e < EE; ++e) cnt[tid * EE + e] = lc[e];
  __syncthreads();

  if (tid < EE) {
    int run = 0;
    for (int j = 0; j < 1024; ++j) {
      const int v = cnt[j * EE + tid];
      cnt[j * EE + tid] = run;
      run += v;
    }
  }
  __syncthreads();

  int off[EE];
#pragma unroll
  for (int e = 0; e < EE; ++e) off[e] = cnt[tid * EE + e];
#pragma unroll
  for (int i = 0; i < 16; ++i) {
    const int f = tid * 16 + i;
    const int e = le[i];
    const int pos = off[e]++;
    const bool keep = pos < CAP;
    const int sl = keep ? (e * CAP + pos) : NSLOT;
    slot[f] = sl;
    if (keep) {
      slot2token[sl] = f >> 1;
      slot2w[sl] = ew[f];
    }
  }
}

// =====================================================================
// K3: gather tokens into bf16 [E*C, D] buffer (zeros for empty slots)
// =====================================================================
__global__ void moe_gather(const float* __restrict__ x,
                           const int* __restrict__ slot2token,
                           bf16* __restrict__ xg) {
  const int sl = blockIdx.x;
  const int t = slot2token[sl];
  const int d = threadIdx.x * 4;
  float4 v = make_float4(0.f, 0.f, 0.f, 0.f);
  if (t >= 0) {
    const int b = t & (BB - 1);
    const int s = t >> 2;
    v = *(const float4*)(x + ((size_t)b * SS + s) * DD + d);
  }
  bf16* o = xg + (size_t)sl * DD + d;
  o[0] = (bf16)v.x; o[1] = (bf16)v.y; o[2] = (bf16)v.z; o[3] = (bf16)v.w;
}

// =====================================================================
// K4/K5: grouped expert GEMM, v_wmma_f32_16x16x32_bf16, double-buffered
// LDS tiles filled by async global->LDS copies.
//   C[m][n] = sum_k A[m][k] * Bt[n][k]
//   block: 8 waves, tile 128(M) x 128(N), K-tile 64 (2 WMMA k-steps).
//   wave tile 32x64 = 2x4 WMMA (12 ds_loads : 8 WMMAs per k-step).
//   ISA 16-bit A/B layout: lane = lo + 16*hi holds row (tile+lo),
//   K chunks [k+8*hi..+7] and [k+16+8*hi..+7].
// =====================================================================
__device__ __forceinline__ v16bf load_frag_lds(const bf16* p, int hi) {
  const v8bf c0 = *(const v8bf*)(p + hi * 8);
  const v8bf c1 = *(const v8bf*)(p + 16 + hi * 8);
  return __builtin_shufflevector(c0, c1, 0, 1, 2, 3, 4, 5, 6, 7,
                                 8, 9, 10, 11, 12, 13, 14, 15);
}

template <bool GELU, typename OutT>
__global__ __launch_bounds__(256)
void moe_gemm_bf16(const bf16* __restrict__ A, const bf16* __restrict__ Bt,
                   OutT* __restrict__ Cout, int Kd, int N,
                   size_t strideA, size_t strideB, size_t strideC) {
  __shared__ bf16 smem[2 * STAGE_ELEMS];   // 72 KB

  const int e = blockIdx.z;
  A    += (size_t)e * strideA;
  Bt   += (size_t)e * strideB;
  Cout += (size_t)e * strideC;

  const int tid = threadIdx.x;
  const int wave = tid >> 5;
  const int lane = tid & 31;
  const int lo = lane & 15, hi = lane >> 4;
  const int wm = wave & 3, wn = wave >> 2;      // 4 x 2 waves
  const int m0 = blockIdx.x * 128;
  const int n0 = blockIdx.y * 128;

  v8f acc[2][4];
#pragma unroll
  for (int i = 0; i < 2; ++i)
#pragma unroll
    for (int j = 0; j < 4; ++j)
#pragma unroll
      for (int r = 0; r < 8; ++r) acc[i][j][r] = 0.f;

  const bf16* Ag = A + (size_t)m0 * Kd;
  const bf16* Bg = Bt + (size_t)n0 * Kd;
  const int ntile = Kd / KTILE;

  // prologue: fill buffer 0
  stage_tile128(Ag, (size_t)Kd, smem, tid);
  stage_tile128(Bg, (size_t)Kd, smem + TILE_ELEMS, tid);
  wait_async_zero();
  __syncthreads();

  for (int kt = 0; kt < ntile; ++kt) {
    const int cur = kt & 1;
    bf16* Ac = smem + cur * STAGE_ELEMS;
    bf16* Bc = Ac + TILE_ELEMS;

    if (kt + 1 < ntile) {  // issue async copies into the other buffer
      bf16* An = smem + (cur ^ 1) * STAGE_ELEMS;
      stage_tile128(Ag + (size_t)(kt + 1) * KTILE, (size_t)Kd, An, tid);
      stage_tile128(Bg + (size_t)(kt + 1) * KTILE, (size_t)Kd, An + TILE_ELEMS, tid);
    }

#pragma unroll
    for (int kk = 0; kk < KTILE; kk += 32) {
      v16bf af[2], bfv[4];
#pragma unroll
      for (int i = 0; i < 2; ++i)
        af[i]  = load_frag_lds(Ac + (wm * 32 + i * 16 + lo) * LDSPITCH + kk, hi);
#pragma unroll
      for (int j = 0; j < 4; ++j)
        bfv[j] = load_frag_lds(Bc + (wn * 64 + j * 16 + lo) * LDSPITCH + kk, hi);
#pragma unroll
      for (int i = 0; i < 2; ++i)
#pragma unroll
        for (int j = 0; j < 4; ++j)
          acc[i][j] = __builtin_amdgcn_wmma_f32_16x16x32_bf16(
              false, af[i], false, bfv[j], (short)0, acc[i][j], false, false);
    }

    wait_async_zero();   // next buffer landed (overlapped with WMMAs above)
    __syncthreads();     // everyone done reading cur, next buffer visible
  }

  // epilogue.  C/D layout: VGPR r -> row (r + 8*hi), lane lo -> col
#pragma unroll
  for (int i = 0; i < 2; ++i)
#pragma unroll
    for (int j = 0; j < 4; ++j) {
      const int col = n0 + wn * 64 + j * 16 + lo;
#pragma unroll
      for (int r = 0; r < 8; ++r) {
        const int row = m0 + wm * 32 + i * 16 + r + 8 * hi;
        float v = acc[i][j][r];
        if (GELU) v = 0.5f * v * (1.f + erff(v * 0.70710678118654752f));
        Cout[(size_t)row * N + col] = (OutT)v;
      }
    }
}

// =====================================================================
// K6: deterministic token-driven weighted combine -> out[B][S][D] fp32
// =====================================================================
__global__ void moe_scatter(const float* __restrict__ y,
                            const int* __restrict__ slot,
                            const float* __restrict__ ew,
                            float* __restrict__ out) {
  const int t = blockIdx.x;
  const int d = threadIdx.x * 4;
  float4 acc = make_float4(0.f, 0.f, 0.f, 0.f);
#pragma unroll
  for (int k = 0; k < KK; ++k) {
    const int sl = slot[t * KK + k];
    if (sl < NSLOT) {
      const float w = ew[t * KK + k];
      const float4 v = *(const float4*)(y + (size_t)sl * DD + d);
      acc.x += w * v.x; acc.y += w * v.y; acc.z += w * v.z; acc.w += w * v.w;
    }
  }
  const int b = t & (BB - 1);
  const int s = t >> 2;
  *(float4*)(out + ((size_t)b * SS + s) * DD + d) = acc;
}

// =====================================================================
extern "C" void kernel_launch(void* const* d_in, const int* in_sizes, int n_in,
                              void* d_out, int out_size, void* d_ws, size_t ws_size,
                              hipStream_t stream) {
  const float* x  = (const float*)d_in[0];   // [B,S,D]
  const float* wr = (const float*)d_in[1];   // [D,E]
  const float* w1 = (const float*)d_in[2];   // [E,D,DFF]
  const float* w2 = (const float*)d_in[3];   // [E,DFF,D]
  float* out = (float*)d_out;                // [B,S,D]

  char* ws = (char*)d_ws;
  size_t off = 0;
  auto take = [&](size_t bytes) { void* p = ws + off; off = (off + bytes + 255) & ~(size_t)255; return p; };

  bf16* w1t = (bf16*)take((size_t)EE * DFF * DD * sizeof(bf16));   // [E,DFF,D]
  bf16* w2t = (bf16*)take((size_t)EE * DD * DFF * sizeof(bf16));   // [E,D,DFF]
  bf16* xg  = (bf16*)take((size_t)NSLOT * DD * sizeof(bf16));      // [E*C,D]
  bf16* h   = (bf16*)take((size_t)NSLOT * DFF * sizeof(bf16));     // [E*C,DFF]
  float* y  = (float*)take((size_t)NSLOT * DD * sizeof(float));    // [E*C,D]
  float* ew = (float*)take((size_t)TT * KK * sizeof(float));
  int* eidx = (int*)take((size_t)TT * KK * sizeof(int));
  int* slot = (int*)take((size_t)TT * KK * sizeof(int));
  int* s2t  = (int*)take((size_t)NSLOT * sizeof(int));
  float* s2w = (float*)take((size_t)NSLOT * sizeof(float));
  (void)ws_size; (void)s2w; (void)in_sizes; (void)n_in; (void)out_size;

  moe_transpose_cast<<<dim3(DFF / 32, DD / 32, EE), dim3(32, 8), 0, stream>>>(w1, w1t, DD, DFF);
  moe_transpose_cast<<<dim3(DD / 32, DFF / 32, EE), dim3(32, 8), 0, stream>>>(w2, w2t, DFF, DD);
  moe_router<<<TT / 8, 256, 0, stream>>>(x, wr, ew, eidx);
  moe_assign<<<1, 1024, 0, stream>>>(eidx, ew, slot, s2t, s2w);
  moe_gather<<<NSLOT, DD / 4, 0, stream>>>(x, s2t, xg);
  // GEMM1 + GeLU: h = gelu(xg @ w1)   M=C, N=DFF, K=D
  moe_gemm_bf16<true, bf16><<<dim3(CAP / 128, DFF / 128, EE), 256, 0, stream>>>(
      xg, w1t, h, DD, DFF,
      (size_t)CAP * DD, (size_t)DFF * DD, (size_t)CAP * DFF);
  // GEMM2: y = h @ w2                 M=C, N=D, K=DFF
  moe_gemm_bf16<false, float><<<dim3(CAP / 128, DD / 128, EE), 256, 0, stream>>>(
      h, w2t, y, DFF, DD,
      (size_t)CAP * DFF, (size_t)DD * DFF, (size_t)CAP * DD);
  moe_scatter<<<TT, DD / 4, 0, stream>>>(y, slot, ew, out);
}
